// PCNN_37864431682335
// MI455X (gfx1250) — matrix operand order repaired and checked by reading, not compiled, not run on hardware
//
#include <hip/hip_runtime.h>
#include <math.h>

// ---------------- types ----------------
typedef __attribute__((ext_vector_type(16))) __bf16 v16bf;
typedef __attribute__((ext_vector_type(8)))  __bf16 v8bf;
typedef __attribute__((ext_vector_type(8)))  float  v8f;
typedef __attribute__((ext_vector_type(4)))  unsigned int v4u;
typedef __attribute__((ext_vector_type(8)))  int v8i;
typedef __attribute__((ext_vector_type(4)))  int v4i;

#define WMMA_BF16(A_, B_, C_) \
  __builtin_amdgcn_wmma_f32_16x16x32_bf16(false, (A_), false, (B_), (short)0, (C_), false, false)

// ---------------- problem constants ----------------
#define BB    128
#define LSEQ  512
#define HL    516      // LSEQ + KH-1 padding rows (zero)
#define DPAD  160      // D=130 padded to multiple of 32
#define WCH   256
#define LP    508
#define KCONV 800      // 5 * 160
#define GG    4
#define LBL   53
#define EPSV  1e-5f
#define NEGM  -100000.0f

// -------- TDM availability hedges (device pass only) --------
#if defined(__AMDGCN__) && __has_builtin(__builtin_amdgcn_tensor_load_to_lds)
#define HAVE_TDM_DEV 1
#endif
#if __has_include(<hip/amd_detail/amd_gfx1250_TDM.h>)
#define TDM_6ARG 1
#endif
#if defined(__AMDGCN__) && __has_builtin(__builtin_amdgcn_s_wait_tensorcnt)
#define HAVE_WAIT_TENSOR 1
#endif

__device__ __forceinline__ __bf16 to_bf16(float f) { return (__bf16)f; }

// ---------------- TDM 2D tile load: contiguous (stride==cols) bf16 tile ----------------
#if defined(HAVE_TDM_DEV)
__device__ __forceinline__ void tdm_load_2d_bf16(const __bf16* gptr, __bf16* lptr,
                                                 unsigned rows, unsigned cols) {
  unsigned long long ga = (unsigned long long)(const void*)gptr;
  unsigned la = (unsigned)(unsigned long long)(void*)lptr; // LDS aperture: low 32 bits = LDS byte addr
  v4u g0;
  g0[0] = 1u;                                            // count=1, user mode, no gather
  g0[1] = la;                                            // lds_addr
  g0[2] = (unsigned)(ga & 0xFFFFFFFFu);                  // global_addr[31:0]
  g0[3] = (unsigned)((ga >> 32) & 0x1FFFFFFu) | (2u << 30); // global_addr[56:32], type=2 (image)
  v8i g1;
  g1[0] = (int)(1u << 16);                               // workgroup_mask=0, data_size=1 (2B)
  g1[1] = (int)((cols & 0xFFFFu) << 16);                 // tensor_dim0[15:0]
  g1[2] = (int)(((cols >> 16) & 0xFFFFu) | ((rows & 0xFFFFu) << 16));  // dim0 hi | dim1 lo
  g1[3] = (int)(((rows >> 16) & 0xFFFFu) | ((cols & 0xFFFFu) << 16));  // dim1 hi | tile_dim0
  g1[4] = (int)(rows & 0xFFFFu);                         // tile_dim1 (tile_dim2=0)
  g1[5] = (int)cols;                                     // tensor_dim0_stride[31:0]
  g1[6] = 0;
  g1[7] = 0;
  v4i z4; 
#pragma unroll
  for (int i = 0; i < 4; ++i) z4[i] = 0;
#if defined(TDM_6ARG)
  v8i z8;
#pragma unroll
  for (int i = 0; i < 8; ++i) z8[i] = 0;
  __builtin_amdgcn_tensor_load_to_lds(g0, g1, z4, z4, z8, 0);
#else
  __builtin_amdgcn_tensor_load_to_lds(g0, g1, z4, z4, 0);
#endif
}
__device__ __forceinline__ void wait_tensor0() {
#if defined(HAVE_WAIT_TENSOR)
  __builtin_amdgcn_s_wait_tensorcnt(0);
#else
  asm volatile("s_wait_tensorcnt 0x0" ::: "memory");
#endif
}
#endif

// Stage a contiguous rows x cols bf16 tile from global into LDS (whole block participates).
__device__ __forceinline__ void stage_tile(const __bf16* g, __bf16* s,
                                           unsigned rows, unsigned cols) {
#if defined(HAVE_TDM_DEV)
  if (threadIdx.x < 32) {          // wave 0 issues one TDM op for the block
    tdm_load_2d_bf16(g, s, rows, cols);
    wait_tensor0();
  }
  __syncthreads();
#else
  unsigned n = rows * cols / 2;    // dwords
  const unsigned* gs = (const unsigned*)g;
  unsigned* ss = (unsigned*)s;
  for (unsigned i = threadIdx.x; i < n; i += 256) ss[i] = gs[i];
  __syncthreads();
#endif
}

// ---------------- fragment loaders ----------------
// A fragment 16x32 bf16, row-major [M][K]; base points at (row0, kb)
__device__ __forceinline__ v16bf ld_frag_a(const __bf16* base, int lda, int lane) {
  const __bf16* p = base + (lane & 15) * lda + ((lane >> 4) << 3);
  v8bf lo = *reinterpret_cast<const v8bf*>(p);
  v8bf hi = *reinterpret_cast<const v8bf*>(p + 16);
  v16bf r;
#pragma unroll
  for (int i = 0; i < 8; ++i) { r[i] = lo[i]; r[i + 8] = hi[i]; }
  return r;
}

// B fragment 32x16 bf16 from W^T stored [N][K] row-major; base points at (col0, kb)
__device__ __forceinline__ v16bf ld_frag_b(const __bf16* base, int ldb, int lane) {
  const __bf16* p = base + (lane & 15) * ldb + ((lane >> 4) << 3);
  v8bf lo = *reinterpret_cast<const v8bf*>(p);
  v8bf hi = *reinterpret_cast<const v8bf*>(p + 16);
  v16bf r;
#pragma unroll
  for (int i = 0; i < 8; ++i) { r[i] = lo[i]; r[i + 8] = hi[i]; }
  return r;
}

// ---------------- prep kernels ----------------
__global__ void k_build_H(const int* __restrict__ x, const float* __restrict__ posEmb,
                          const float* __restrict__ emb, __bf16* __restrict__ H) {
  int idx = blockIdx.x * 256 + threadIdx.x;
  if (idx >= BB * HL * DPAD) return;
  int b = idx / (HL * DPAD);
  int r = idx - b * (HL * DPAD);
  int l = r / DPAD;
  int d = r - l * DPAD;
  float v = 0.0f;
  if (l < LSEQ) {
    if (d < 128)      v = emb[(size_t)x[b * LSEQ + l] * 128 + d];
    else if (d < 130) v = posEmb[((size_t)(b * LSEQ + l)) * 2 + (d - 128)];
  }
  H[idx] = to_bf16(v);
}

__global__ void k_prep_wt1(const float* __restrict__ w, __bf16* __restrict__ Wt1) {
  int idx = blockIdx.x * 256 + threadIdx.x;
  if (idx >= WCH * KCONV) return;
  int o = idx / KCONV;
  int k = idx - o * KCONV;
  int kh = k / DPAD;
  int d  = k - kh * DPAD;
  float v = (d < 130) ? w[(o * 5 + kh) * 130 + d] : 0.0f;
  Wt1[idx] = to_bf16(v);
}

__global__ void k_f32_to_bf16(const float* __restrict__ s, __bf16* __restrict__ d, int n) {
  int idx = blockIdx.x * 256 + threadIdx.x;
  if (idx < n) d[idx] = to_bf16(s[idx]);
}

// ---------------- conv1 GEMM: M=B*512, N=256, K=800; fused bias+BN+PReLU ----------------
// A panel (68 rows x 160) staged in LDS via TDM; B fragments software-pipelined from global.
__global__ void k_conv_gemm(const __bf16* __restrict__ H, const __bf16* __restrict__ Wt,
                            const float* __restrict__ bias, const float* __restrict__ g,
                            const float* __restrict__ bb, const float* __restrict__ mn,
                            const float* __restrict__ vr, const float* __restrict__ pw,
                            float* __restrict__ Y0, __bf16* __restrict__ Ybf) {
  __shared__ __bf16 As[68 * DPAD];               // 21.25 KB
  const int lane  = threadIdx.x & 31;
  const int wave  = (blockIdx.x << 3) + (threadIdx.x >> 5);
  const int ntile = wave & 15;
  const int mg    = wave >> 4;       // 0..1023 (same for all 8 waves in a block)
  const int row0  = mg << 6;         // 64-row block (inside one batch: 512%64==0)
  const int b     = row0 >> 9;
  const int t0    = row0 & 511;
  const __bf16* Wb = Wt + (size_t)(ntile << 4) * KCONV;

  stage_tile(H + ((size_t)b * HL + t0) * DPAD, As, 68, DPAD);

  v8f acc[4];
#pragma unroll
  for (int mi = 0; mi < 4; ++mi)
#pragma unroll
    for (int e = 0; e < 8; ++e) acc[mi][e] = 0.0f;

  v16bf fb = ld_frag_b(Wb, KCONV, lane);
  for (int kb = 0; kb < KCONV; kb += 32) {
    v16bf fbn = fb;
    if (kb + 32 < KCONV) fbn = ld_frag_b(Wb + kb + 32, KCONV, lane);  // prefetch next B frag
    int kh = kb / DPAD;           // 160 % 32 == 0 -> k-block never straddles kh
    int db = kb - kh * DPAD;
    v16bf fa[4];
#pragma unroll
    for (int mi = 0; mi < 4; ++mi)
      fa[mi] = ld_frag_a(As + ((mi << 4) + kh) * DPAD + db, DPAD, lane);  // ds_load
#pragma unroll
    for (int mi = 0; mi < 4; ++mi)
      acc[mi] = WMMA_BF16(fa[mi], fb, acc[mi]);
    fb = fbn;
  }

  const int n   = (ntile << 4) + (lane & 15);
  const float sc = g[n] * rsqrtf(vr[n] + EPSV);
  const float sh = bb[n] - mn[n] * sc;
  const float bs = bias[n];
  const float p  = pw[0];
  const int mrow = (lane >> 4) << 3;
#pragma unroll
  for (int mi = 0; mi < 4; ++mi) {
#pragma unroll
    for (int e = 0; e < 8; ++e) {
      int grow = row0 + (mi << 4) + mrow + e;
      float v = acc[mi][e] + bs;
      v = v * sc + sh;
      v = (v > 0.0f) ? v : p * v;
      size_t o = ((size_t)grow << 8) + n;
      Y0[o]  = v;
      Ybf[o] = to_bf16(v);
    }
  }
}

// ---------------- channel-block GEMM: M=B*512, N=256, K=256; fused bias ----------------
__global__ void k_cb_gemm(const __bf16* __restrict__ A, const __bf16* __restrict__ Wt,
                          const float* __restrict__ bias, float* __restrict__ Z) {
  __shared__ __bf16 As[64 * WCH];                // 32 KB
  const int lane  = threadIdx.x & 31;
  const int wave  = (blockIdx.x << 3) + (threadIdx.x >> 5);
  const int ntile = wave & 15;
  const int mg    = wave >> 4;
  const int row0  = mg << 6;
  const __bf16* Wb = Wt + (size_t)(ntile << 4) * WCH;

  stage_tile(A + (size_t)row0 * WCH, As, 64, WCH);

  v8f acc[4];
#pragma unroll
  for (int mi = 0; mi < 4; ++mi)
#pragma unroll
    for (int e = 0; e < 8; ++e) acc[mi][e] = 0.0f;

  v16bf fb = ld_frag_b(Wb, WCH, lane);
#pragma unroll
  for (int kb = 0; kb < WCH; kb += 32) {
    v16bf fbn = fb;
    if (kb + 32 < WCH) fbn = ld_frag_b(Wb + kb + 32, WCH, lane);      // prefetch next B frag
    v16bf fa[4];
#pragma unroll
    for (int mi = 0; mi < 4; ++mi)
      fa[mi] = ld_frag_a(As + (mi << 4) * WCH + kb, WCH, lane);       // ds_load
#pragma unroll
    for (int mi = 0; mi < 4; ++mi)
      acc[mi] = WMMA_BF16(fa[mi], fb, acc[mi]);
    fb = fbn;
  }

  const int n  = (ntile << 4) + (lane & 15);
  const float bs = bias[n];
  const int mrow = (lane >> 4) << 3;
#pragma unroll
  for (int mi = 0; mi < 4; ++mi) {
#pragma unroll
    for (int e = 0; e < 8; ++e) {
      int grow = row0 + (mi << 4) + mrow + e;
      Z[((size_t)grow << 8) + n] = acc[mi][e] + bs;
    }
  }
}

// ---------------- GroupNorm stats: one block per (b,g), t<508 only ----------------
__global__ void k_gn_stats(const float* __restrict__ Z, float* __restrict__ stats) {
  __shared__ float s1[256], s2[256];
  const int bg = blockIdx.x;
  const int b  = bg >> 2;
  const int gi = bg & 3;
  const float* base = Z + (((size_t)b << 9) << 8) + (gi << 6);
  float sm = 0.0f, ss = 0.0f;
  for (int idx = threadIdx.x; idx < LP * 64; idx += 256) {
    int t = idx >> 6;
    int c = idx & 63;
    float v = base[((size_t)t << 8) + c];
    sm += v; ss += v * v;
  }
  s1[threadIdx.x] = sm; s2[threadIdx.x] = ss;
  __syncthreads();
  for (int s = 128; s > 0; s >>= 1) {
    if (threadIdx.x < s) { s1[threadIdx.x] += s1[threadIdx.x + s]; s2[threadIdx.x] += s2[threadIdx.x + s]; }
    __syncthreads();
  }
  if (threadIdx.x == 0) {
    float inv = 1.0f / (float)(LP * 64);
    float mu  = s1[0] * inv;
    float var = s2[0] * inv - mu * mu;
    stats[bg * 2]     = mu;
    stats[bg * 2 + 1] = rsqrtf(var + EPSV);
  }
}

// ---------------- GN apply + PReLU + residual (in place on Z) ----------------
__global__ void k_gn_apply(float* __restrict__ Z, const float* __restrict__ Yc,
                           const float* __restrict__ gam, const float* __restrict__ bet,
                           const float* __restrict__ pr, const float* __restrict__ stats,
                           __bf16* __restrict__ Ybf) {
  size_t idx = (size_t)blockIdx.x * 256 + threadIdx.x;   // < 128*512*256
  int n  = (int)(idx & 255);
  size_t bt = idx >> 8;
  int b  = (int)(bt >> 9);
  int gi = n >> 6;
  float mu = stats[(b * GG + gi) * 2];
  float rs = stats[(b * GG + gi) * 2 + 1];
  float v = (Z[idx] - mu) * rs * gam[n] + bet[n];
  float p = pr[0];
  v = (v > 0.0f) ? v : p * v;
  float y = v + Yc[idx];
  Z[idx]  = y;
  Ybf[idx] = to_bf16(y);
}

// ---------------- segment max (y+xI with additive -1e5 masks) ----------------
__global__ void k_segmax(const float* __restrict__ Yc, const float* __restrict__ Y0,
                         const int* __restrict__ pos, float* __restrict__ feat,
                         __bf16* __restrict__ featbf) {
  const int b = blockIdx.x;
  const int w = threadIdx.x;
  const int p0 = pos[b * 2] + 1;
  const int p1 = pos[b * 2 + 1] + 1;
  float a1 = -3.4e38f, a2 = -3.4e38f, a3 = -3.4e38f;
  const size_t base = ((size_t)b << 9) << 8;
  for (int t = 0; t < LP; ++t) {
    size_t o = base + ((size_t)t << 8) + w;
    float v = Yc[o] + Y0[o];
    float m1 = (t <  p0)           ? 0.0f : NEGM;
    float m2 = (t >= p0 && t < p1) ? 0.0f : NEGM;
    float m3 = (t >= p1)           ? 0.0f : NEGM;
    a1 = fmaxf(a1, v + m1);
    a2 = fmaxf(a2, v + m2);
    a3 = fmaxf(a3, v + m3);
  }
  int fo = b * 768 + w;
  feat[fo]       = a1;  featbf[fo]       = to_bf16(a1);
  feat[fo + 256] = a2;  featbf[fo + 256] = to_bf16(a2);
  feat[fo + 512] = a3;  featbf[fo + 512] = to_bf16(a3);
}

// ---------------- dense GEMM: M=128, N=256, runtime K; fused bias+BN+PReLU ----------------
__global__ void k_dense_gemm(const __bf16* __restrict__ A, int K, const __bf16* __restrict__ Wt,
                             const float* __restrict__ bias, const float* __restrict__ g,
                             const float* __restrict__ bb, const float* __restrict__ mn,
                             const float* __restrict__ vr, const float* __restrict__ pw,
                             float* __restrict__ outF, __bf16* __restrict__ outBf) {
  const int lane  = threadIdx.x & 31;
  const int wave  = (blockIdx.x << 3) + (threadIdx.x >> 5);  // 0..31
  const int ntile = wave & 15;
  const int mg    = wave >> 4;     // 0..1
  const int row0  = mg << 6;
  const __bf16* Wb = Wt + (size_t)(ntile << 4) * K;

  v8f acc[4];
#pragma unroll
  for (int mi = 0; mi < 4; ++mi)
#pragma unroll
    for (int e = 0; e < 8; ++e) acc[mi][e] = 0.0f;

  for (int kb = 0; kb < K; kb += 32) {
    v16bf fb = ld_frag_b(Wb + kb, K, lane);
#pragma unroll
    for (int mi = 0; mi < 4; ++mi) {
      v16bf fa = ld_frag_a(A + (size_t)(row0 + (mi << 4)) * K + kb, K, lane);
      acc[mi] = WMMA_BF16(fa, fb, acc[mi]);
    }
  }

  const int n   = (ntile << 4) + (lane & 15);
  const float sc = g[n] * rsqrtf(vr[n] + EPSV);
  const float sh = bb[n] - mn[n] * sc;
  const float bs = bias[n];
  const float p  = pw[0];
  const int mrow = (lane >> 4) << 3;
#pragma unroll
  for (int mi = 0; mi < 4; ++mi) {
#pragma unroll
    for (int e = 0; e < 8; ++e) {
      int grow = row0 + (mi << 4) + mrow + e;
      float v = acc[mi][e] + bs;
      v = v * sc + sh;
      v = (v > 0.0f) ? v : p * v;
      int o = (grow << 8) + n;
      outF[o] = v;
      if (outBf) outBf[o] = to_bf16(v);
    }
  }
}

// ---------------- final logits + softmax ----------------
__global__ void k_final(const float* __restrict__ h1, const float* __restrict__ fw,
                        const float* __restrict__ fb, float* __restrict__ out) {
  __shared__ float sd[64];
  const int b = blockIdx.x;
  const int t = threadIdx.x;
  float l = -3.4e38f;
  if (t < LBL) {
    float s = fb[t];
    const float* hr = h1 + b * 256;
    const float* wr = fw + t * 256;
    for (int k = 0; k < 256; ++k) s += hr[k] * wr[k];
    l = s;
  }
  sd[t] = l; __syncthreads();
  for (int s = 32; s > 0; s >>= 1) { if (t < s) sd[t] = fmaxf(sd[t], sd[t + s]); __syncthreads(); }
  float mx = sd[0]; __syncthreads();
  float e = (t < LBL) ? expf(l - mx) : 0.0f;
  sd[t] = e; __syncthreads();
  for (int s = 32; s > 0; s >>= 1) { if (t < s) sd[t] += sd[t + s]; __syncthreads(); }
  float sum = sd[0];
  if (t < LBL) out[b * LBL + t] = e / sum;
}

// ---------------- host ----------------
extern "C" void kernel_launch(void* const* d_in, const int* in_sizes, int n_in,
                              void* d_out, int out_size, void* d_ws, size_t ws_size,
                              hipStream_t stream) {
  const int*   x      = (const int*)d_in[0];
  const int*   posE   = (const int*)d_in[1];
  const float* posEmb = (const float*)d_in[2];
  const float* emb    = (const float*)d_in[3];
  const float* conv1w = (const float*)d_in[4];
  const float* conv1b = (const float*)d_in[5];
  const float* bn1g   = (const float*)d_in[6];
  const float* bn1b   = (const float*)d_in[7];
  const float* bn1m   = (const float*)d_in[8];
  const float* bn1v   = (const float*)d_in[9];
  const float* prelu1 = (const float*)d_in[10];
  const float* cbw_f  = (const float*)d_in[11];
  const float* cbb    = (const float*)d_in[12];
  const float* gng    = (const float*)d_in[13];
  const float* gnb    = (const float*)d_in[14];
  const float* cbpr   = (const float*)d_in[15];
  const float* d0w_f  = (const float*)d_in[16];
  const float* d0b    = (const float*)d_in[17];
  const float* d1w_f  = (const float*)d_in[18];
  const float* d1b    = (const float*)d_in[19];
  const float* bndg   = (const float*)d_in[20];
  const float* bndb   = (const float*)d_in[21];
  const float* bndm   = (const float*)d_in[22];
  const float* bndv   = (const float*)d_in[23];
  const float* prd    = (const float*)d_in[24];
  const float* finw   = (const float*)d_in[25];
  const float* finb   = (const float*)d_in[26];
  (void)in_sizes; (void)n_in; (void)out_size; (void)ws_size;

  char* ws = (char*)d_ws;
  size_t off = 0;
  auto alloc = [&](size_t bytes) -> void* {
    void* p = ws + off;
    off += (bytes + 255) & ~(size_t)255;
    return p;
  };

  __bf16* H      = (__bf16*)alloc((size_t)BB * HL * DPAD * 2);
  __bf16* Wt1    = (__bf16*)alloc((size_t)WCH * KCONV * 2);
  __bf16* cbw    = (__bf16*)alloc((size_t)4 * WCH * WCH * 2);
  __bf16* d0w    = (__bf16*)alloc((size_t)WCH * 768 * 2);
  __bf16* d1w    = (__bf16*)alloc((size_t)WCH * WCH * 2);
  float*  Y0     = (float*) alloc((size_t)BB * LSEQ * WCH * 4);
  float*  Pb     = (float*) alloc((size_t)BB * LSEQ * WCH * 4);
  float*  Qb     = (float*) alloc((size_t)BB * LSEQ * WCH * 4);
  __bf16* Ybf    = (__bf16*)alloc((size_t)BB * LSEQ * WCH * 2);
  float*  stats  = (float*) alloc((size_t)BB * GG * 2 * 4);
  float*  feat   = (float*) alloc((size_t)BB * 768 * 4);
  __bf16* featbf = (__bf16*)alloc((size_t)BB * 768 * 2);
  float*  h0     = (float*) alloc((size_t)BB * WCH * 4);
  __bf16* h0bf   = (__bf16*)alloc((size_t)BB * WCH * 2);
  float*  h1     = (float*) alloc((size_t)BB * WCH * 4);

  // ---- prep ----
  {
    int n = BB * HL * DPAD;
    k_build_H<<<(n + 255) / 256, 256, 0, stream>>>(x, posEmb, emb, H);
  }
  k_prep_wt1<<<(WCH * KCONV + 255) / 256, 256, 0, stream>>>(conv1w, Wt1);
  k_f32_to_bf16<<<(4 * WCH * WCH + 255) / 256, 256, 0, stream>>>(cbw_f, cbw, 4 * WCH * WCH);
  k_f32_to_bf16<<<(WCH * 768 + 255) / 256, 256, 0, stream>>>(d0w_f, d0w, WCH * 768);
  k_f32_to_bf16<<<(WCH * WCH + 255) / 256, 256, 0, stream>>>(d1w_f, d1w, WCH * WCH);

  // ---- conv1 + BN + PReLU (WMMA, TDM-staged A) ----
  k_conv_gemm<<<2048, 256, 0, stream>>>(H, Wt1, conv1b, bn1g, bn1b, bn1m, bn1v, prelu1, Y0, Ybf);

  // ---- 4 residual channel blocks ----
  for (int i = 0; i < 4; ++i) {
    float* Z  = (i & 1) ? Qb : Pb;
    const float* Yc = (i == 0) ? Y0 : ((i & 1) ? Pb : Qb);
    k_cb_gemm<<<2048, 256, 0, stream>>>(Ybf, cbw + (size_t)i * WCH * WCH, cbb + i * WCH, Z);
    k_gn_stats<<<BB * GG, 256, 0, stream>>>(Z, stats);
    k_gn_apply<<<65536, 256, 0, stream>>>(Z, Yc, gng + i * WCH, gnb + i * WCH,
                                          cbpr + i, stats, Ybf);
  }

  // ---- segment max over (Y4 + Y0) ----
  k_segmax<<<BB, 256, 0, stream>>>(Qb, Y0, posE, feat, featbf);

  // ---- dense head (WMMA) ----
  k_dense_gemm<<<4, 256, 0, stream>>>(featbf, 768, d0w, d0b, bndg, bndb, bndm, bndv, prd, h0, h0bf);
  k_dense_gemm<<<4, 256, 0, stream>>>(h0bf, 256, d1w, d1b, bndg + WCH, bndb + WCH,
                                      bndm + WCH, bndv + WCH, prd + 1, h1, (__bf16*)nullptr);

  // ---- final logits + softmax ----
  k_final<<<BB, 64, 0, stream>>>(h1, finw, finb, (float*)d_out);
}
